// PersLay_62689342652499
// MI455X (gfx1250) — compile-verified, matrix operand order checked by reading, not run on hardware
//
#include <hip/hip_runtime.h>
#include <math.h>

typedef __attribute__((ext_vector_type(16))) _Float16 v16h;
typedef __attribute__((ext_vector_type(8)))  float    v8f;

#define T_PTS 524288
#define B_GR  1024
#define Q_C   64
#define GF_N  16
#define WW_N  64
#define WF_N  128
#define NC_N  10

// ---------------------------------------------------------------- zero scratch
__global__ void zero_kernel(float* __restrict__ p, int n) {
  int i = blockIdx.x * blockDim.x + threadIdx.x;
  if (i < n) p[i] = 0.0f;
}

// ------------------------------------------- repack Ww1 (f32 [64,64]) -> f16 B-layout
// B operand for (n_tile, kc): lane l holds column n = n_tile*16 + (l&15),
// 16 halves j=0..15 with k_local = j + (j<8?0:8) + (l>=16?8:0), k = kc*32+k_local.
__global__ void prep_w1(const float* __restrict__ Ww1, _Float16* __restrict__ out) {
  int tid  = threadIdx.x;          // 0..255
  int grp  = tid >> 5;             // n_tile*2 + kc  (0..7)
  int lane = tid & 31;
  int n_tile = grp >> 1, kc = grp & 1;
  int hi = lane >> 4;
  int n  = n_tile * 16 + (lane & 15);
  for (int j = 0; j < 16; ++j) {
    int kl = j + ((j < 8) ? 0 : 8) + (hi ? 8 : 0);
    int k  = kc * 32 + kl;
    out[(size_t)(grp * 32 + lane) * 16 + j] = (_Float16)Ww1[k * WW_N + n];
  }
}

// ---------------------------------------------------------------- branch kernel
// One wave = one 16-point tile. 8 waves / block, grid.y selects diagram 0/1.
__global__ __launch_bounds__(256)
void perslay_branch(const float* __restrict__ diag0, const int* __restrict__ batch0,
                    const float* __restrict__ diag1, const int* __restrict__ batch1,
                    const float* __restrict__ theta,
                    const float* __restrict__ Ww0, const float* __restrict__ bw0,
                    const float* __restrict__ bw1,
                    const float* __restrict__ Wwo, const float* __restrict__ bwo,
                    const _Float16* __restrict__ W1h,
                    float* __restrict__ v0, float* __restrict__ v1) {
  const float* diag;  const int* batch;  float* vout;
  if (blockIdx.y == 0) { diag = diag0; batch = batch0; vout = v0; }
  else                 { diag = diag1; batch = batch1; vout = v1; }

  __shared__ float sW0[2 * WW_N];
  __shared__ float sb0[WW_N], sWo[WW_N], sb1[WW_N];
  __shared__ float sTh[2 * Q_C];
  __shared__ float sbo_s;
  __shared__ float s_pts[8][16][2];
  __shared__ int   s_bt[8][16];

  int tid = threadIdx.x;
  for (int i = tid; i < 2 * WW_N; i += 256) sW0[i] = Ww0[i];
  for (int i = tid; i < WW_N;    i += 256) { sb0[i] = bw0[i]; sWo[i] = Wwo[i]; sb1[i] = bw1[i]; }
  for (int i = tid; i < 2 * Q_C; i += 256) sTh[i] = theta[i];
  if (tid == 0) sbo_s = bwo[0];

  int wv   = tid >> 5;
  int lane = tid & 31;
  int hi   = lane >> 4;
  int m16  = lane & 15;
  int base = (blockIdx.x * 8 + wv) * 16;     // first point of this wave's tile

  if (lane < 16) {
    s_pts[wv][lane][0] = diag[(size_t)(base + lane) * 2 + 0];
    s_pts[wv][lane][1] = diag[(size_t)(base + lane) * 2 + 1];
    s_bt[wv][lane]     = batch[base + lane];
  }
  __syncthreads();

  // A-operand rows: row M = lane%16 (both halves share M, differ in K set)
  float px = s_pts[wv][m16][0];
  float py = s_pts[wv][m16][1];

  // layer 0 (K=2) computed on VALU, written straight into the WMMA A layout (f16)
  v16h A[2];
  for (int kc = 0; kc < 2; ++kc) {
    for (int j = 0; j < 16; ++j) {
      int kl = j + ((j < 8) ? 0 : 8) + (hi ? 8 : 0);
      int k  = kc * 32 + kl;
      float h = fmaf(px, sW0[k], fmaf(py, sW0[WW_N + k], sb0[k]));
      A[kc][j] = (_Float16)fmaxf(h, 0.0f);
    }
  }

  // h1 = relu(h0 @ Ww1 + b1) via WMMA; fold immediately into w-logit = h1 . Wwo
  float wl[8];
  for (int r = 0; r < 8; ++r) wl[r] = 0.0f;
  for (int n = 0; n < 4; ++n) {
    v8f c = {};
    for (int kc = 0; kc < 2; ++kc) {
      v16h Bv = *(const v16h*)(W1h + (size_t)((n * 2 + kc) * 32 + lane) * 16);
      c = __builtin_amdgcn_wmma_f32_16x16x32_f16(false, A[kc], false, Bv,
                                                 (short)0, c, false, false);
    }
    int q = n * 16 + m16;                    // C layout: this lane's N column
    float b1q = sb1[q], woq = sWo[q];
    for (int r = 0; r < 8; ++r)
      wl[r] += fmaxf(c[r] + b1q, 0.0f) * woq;
  }
  // reduce the dot over the 16-lane N-group (xor 1,2,4,8 stays in the half)
  for (int off = 1; off < 16; off <<= 1)
    for (int r = 0; r < 8; ++r)
      wl[r] += __shfl_xor(wl[r], off, 32);

  float wgt[8];                              // wgt[r] = sigmoid weight for M = r + 8*hi
  for (int r = 0; r < 8; ++r)
    wgt[r] = 1.0f / (1.0f + __expf(-(wl[r] + sbo_s)));

  // Gaussian rep * weight, then segment sum (sorted batch ids -> mostly uniform tiles)
  const float inv2s2 = 50.0f;                // 1 / (2 * 0.1^2)
  bool uniform = (s_bt[wv][0] == s_bt[wv][15]);
  int  b0 = s_bt[wv][0];
  for (int n = 0; n < 4; ++n) {
    int q = n * 16 + m16;
    float tx = sTh[q * 2 + 0], ty = sTh[q * 2 + 1];
    float acc = 0.0f;
    for (int r = 0; r < 8; ++r) {
      int M = r + 8 * hi;
      float dx = s_pts[wv][M][0] - tx;
      float dy = s_pts[wv][M][1] - ty;
      float val = wgt[r] * __expf(-(dx * dx + dy * dy) * inv2s2);
      if (uniform) acc += val;
      else atomicAdd(&vout[(size_t)s_bt[wv][M] * Q_C + q], val);
    }
    if (uniform) {
      acc += __shfl_xor(acc, 16, 32);        // combine M halves (same q in both)
      if (!hi) atomicAdd(&vout[(size_t)b0 * Q_C + q], acc);
    }
  }
}

// ---------------------------------------------------------------- head MLP (tiny)
__global__ __launch_bounds__(128)
void head_mlp(const float* __restrict__ v0, const float* __restrict__ v1,
              const float* __restrict__ gf,
              const float* __restrict__ Wr0, const float* __restrict__ br0,
              const float* __restrict__ Wr1, const float* __restrict__ br1,
              const float* __restrict__ Wro, const float* __restrict__ bro,
              float* __restrict__ out) {
  __shared__ float xs[2 * Q_C + GF_N];       // 144
  __shared__ float h0s[WF_N], h1s[WF_N];
  int g = blockIdx.x, t = threadIdx.x;
  for (int i = t; i < 2 * Q_C + GF_N; i += 128)
    xs[i] = (i < Q_C)     ? v0[(size_t)g * Q_C + i]
          : (i < 2 * Q_C) ? v1[(size_t)g * Q_C + (i - Q_C)]
                          : gf[(size_t)g * GF_N + (i - 2 * Q_C)];
  __syncthreads();
  float a = br0[t];
  for (int k = 0; k < 2 * Q_C + GF_N; ++k) a = fmaf(xs[k], Wr0[k * WF_N + t], a);
  h0s[t] = fmaxf(a, 0.0f);
  __syncthreads();
  float b = br1[t];
  for (int k = 0; k < WF_N; ++k) b = fmaf(h0s[k], Wr1[k * WF_N + t], b);
  h1s[t] = fmaxf(b, 0.0f);
  __syncthreads();
  if (t < NC_N) {
    float o = bro[t];
    for (int k = 0; k < WF_N; ++k) o = fmaf(h1s[k], Wro[k * NC_N + t], o);
    out[(size_t)g * NC_N + t] = o;
  }
}

// ---------------------------------------------------------------- launch
extern "C" void kernel_launch(void* const* d_in, const int* in_sizes, int n_in,
                              void* d_out, int out_size, void* d_ws, size_t ws_size,
                              hipStream_t stream) {
  const float* diag0 = (const float*)d_in[0];
  const float* diag1 = (const float*)d_in[1];
  const float* gfeat = (const float*)d_in[2];
  const float* theta = (const float*)d_in[3];
  const float* Ww0   = (const float*)d_in[4];
  const float* bw0   = (const float*)d_in[5];
  const float* Ww1   = (const float*)d_in[6];
  const float* bw1   = (const float*)d_in[7];
  const float* Wwo   = (const float*)d_in[8];
  const float* bwo   = (const float*)d_in[9];
  const float* Wr0   = (const float*)d_in[10];
  const float* br0   = (const float*)d_in[11];
  const float* Wr1   = (const float*)d_in[12];
  const float* br1   = (const float*)d_in[13];
  const float* Wro   = (const float*)d_in[14];
  const float* bro   = (const float*)d_in[15];
  const int*   bat0  = (const int*)d_in[16];
  const int*   bat1  = (const int*)d_in[17];
  float*       out   = (float*)d_out;

  // workspace layout: v0 [B*Q f32] | v1 [B*Q f32] | W1h [4096 f16]
  float*     ws_v0 = (float*)d_ws;
  float*     ws_v1 = ws_v0 + (size_t)B_GR * Q_C;
  _Float16*  W1h   = (_Float16*)((char*)d_ws + (size_t)2 * B_GR * Q_C * sizeof(float));

  // 1) zero the pooled accumulators (atomics target)
  int nzero = 2 * B_GR * Q_C;
  zero_kernel<<<(nzero + 255) / 256, 256, 0, stream>>>(ws_v0, nzero);

  // 2) repack Ww1 into f16 WMMA B layout
  prep_w1<<<1, 256, 0, stream>>>(Ww1, W1h);

  // 3) both PersLay branches: (T/128 blocks) x (2 branches), 8 waves x 16 pts each
  dim3 grid(T_PTS / 128, 2, 1);
  perslay_branch<<<grid, 256, 0, stream>>>(diag0, bat0, diag1, bat1, theta,
                                           Ww0, bw0, bw1, Wwo, bwo, W1h,
                                           ws_v0, ws_v1);

  // 4) final classifier head
  head_mlp<<<B_GR, 128, 0, stream>>>(ws_v0, ws_v1, gfeat,
                                     Wr0, br0, Wr1, br1, Wro, bro, out);
}
